// TemporalHeteroHG_61314953117927
// MI455X (gfx1250) — matrix operand order repaired and checked by reading, not compiled, not run on hardware
//
#include <hip/hip_runtime.h>
#include <math.h>

// ---------------------------------------------------------------------------
// TemporalHeteroHG for MI455X (gfx1250, wave32, WMMA).
// N=10000, E=160000, C_IN=64, HID=64, C_OUT=32, FC_HID=128.
//
// Input flattening assumed to follow setup_inputs() dict insertion order:
//  0:x 1:node_ts 2:edge_index_a 3:edge_index_b
//  4:Wih0 5:Whh0 6:bih0 7:bhh0 8:bias0
//  9:Wih1 10:Whh1 11:bih1 12:bhh1 13:bias1
//  14:W1 15:b1 16:W2 17:b2
// ---------------------------------------------------------------------------

typedef __attribute__((ext_vector_type(16))) _Float16 v16h;
typedef __attribute__((ext_vector_type(4)))  _Float16 v4h;
typedef __attribute__((ext_vector_type(8)))  float    v8f;

#define FILLV (-999.0f)

// Native transcendental path: gfx1250 has v_tanh_f32 (TRANS). Guarded; the
// fallback uses v_exp_f32 + v_rcp_f32 and saturates correctly for |x| huge
// (fill rows drive |z| ~ 1e5): exp underflow -> -1, rcp(inf)=0 -> +1.
#if defined(__has_builtin)
#if __has_builtin(__builtin_amdgcn_tanhf)
#define HAVE_NATIVE_TANH 1
#endif
#endif

__device__ __forceinline__ float tanh_(float x) {
#ifdef HAVE_NATIVE_TANH
  return __builtin_amdgcn_tanhf(x);
#else
  return 1.0f - 2.0f * __builtin_amdgcn_rcpf(__expf(2.0f * x) + 1.0f);
#endif
}
__device__ __forceinline__ float sig_(float x) {
  return fmaf(tanh_(0.5f * x), 0.5f, 0.5f);
}

// ------------------------------ prep kernels -------------------------------

__global__ void k_zero2(int* __restrict__ p0, int* __restrict__ p1, int n) {
  int i = blockIdx.x * blockDim.x + threadIdx.x;
  int* p = blockIdx.y ? p1 : p0;
  if (i < n) p[i] = 0;
}

// histogram of eligible edges per destination (atomic add: order-independent)
__global__ void k_count2(const int* __restrict__ e0, const int* __restrict__ e1,
                         const float* __restrict__ ts, int* __restrict__ c0,
                         int* __restrict__ c1, int E) {
  int e = blockIdx.x * blockDim.x + threadIdx.x;
  if (e >= E) return;
  const int* ei = blockIdx.y ? e1 : e0;
  int* cnt = blockIdx.y ? c1 : c0;
  int frm = ei[e], to = ei[E + e];
  if (ts[frm] <= ts[to]) atomicAdd(&cnt[to], 1);
}

// single-block exclusive scan (starts) + max degree; blockIdx.x = edge set
__global__ void k_scan2(const int* __restrict__ c0, const int* __restrict__ c1,
                        int* __restrict__ s0, int* __restrict__ s1,
                        int* __restrict__ m0, int* __restrict__ m1, int n) {
  const int* cnt = blockIdx.x ? c1 : c0;
  int* starts = blockIdx.x ? s1 : s0;
  int* mdeg = blockIdx.x ? m1 : m0;
  __shared__ int buf[256];
  __shared__ int carry;
  __shared__ int mx;
  int tid = threadIdx.x;
  if (tid == 0) { carry = 0; mx = 0; }
  __syncthreads();
  for (int base = 0; base < n; base += 256) {
    int v = (base + tid < n) ? cnt[base + tid] : 0;
    atomicMax(&mx, v);
    buf[tid] = v;
    __syncthreads();
    for (int off = 1; off < 256; off <<= 1) {
      int t = (tid >= off) ? buf[tid - off] : 0;
      __syncthreads();
      buf[tid] += t;
      __syncthreads();
    }
    if (base + tid < n) starts[base + tid] = carry + buf[tid] - v;
    __syncthreads();
    if (tid == 0) carry += buf[255];
    __syncthreads();
  }
  if (tid == 0) *mdeg = mx;
}

// deterministic, order-stable CSR fill: one thread per destination node scans
// the edge list in original order (reproduces stable argsort-by-dest).
// Edge list (1.3 MB) is L2-resident; loads are wave-uniform per iteration.
__global__ void k_fill2(const int* __restrict__ e0, const int* __restrict__ e1,
                        const float* __restrict__ ts, const int* __restrict__ s0,
                        const int* __restrict__ s1, int* __restrict__ f0,
                        int* __restrict__ f1, int nnodes, int E) {
  int node = blockIdx.x * blockDim.x + threadIdx.x;
  if (node >= nnodes) return;
  const int* ei = blockIdx.y ? e1 : e0;
  const int* starts = blockIdx.y ? s1 : s0;
  int* sfrm = blockIdx.y ? f1 : f0;
  int w = starts[node];
  float tnode = ts[node];
  for (int e = 0; e < E; ++e) {
    int to = ei[E + e];
    if (to == node) {
      int frm = ei[e];
      if (ts[frm] <= tnode) sfrm[w++] = frm;
    }
  }
}

// f32 -> f16 feature pre-conversion (once, outside the T-loop)
__global__ void k_f32_to_f16(const float* __restrict__ in,
                             _Float16* __restrict__ out, int n4) {
  int i = blockIdx.x * blockDim.x + threadIdx.x;
  if (i >= n4) return;
  float4 v = ((const float4*)in)[i];
  v4h h = {(_Float16)v.x, (_Float16)v.y, (_Float16)v.z, (_Float16)v.w};
  ((v4h*)out)[i] = h;
}

// ------------------------------ LSTM conv ----------------------------------
// One 256-thread block (8 wave32) per 16-node tile; blockIdx.y = edge set.
// Gates Z = [16 x 4H] via v_wmma_f32_16x16x32_f16; B operands (Wih^T / Whh^T
// tiles) are packed into registers ONCE (loop-invariant over the T steps).

template <int CIN, int H>
__global__ __launch_bounds__(256) void k_lstm(
    const _Float16* __restrict__ xsrc, const int* __restrict__ sfrm0,
    const int* __restrict__ sfrm1, const int* __restrict__ st0,
    const int* __restrict__ st1, const int* __restrict__ cnt0,
    const int* __restrict__ cnt1, const int* __restrict__ md0,
    const int* __restrict__ md1, const float* __restrict__ Wih,
    const float* __restrict__ Whh, const float* __restrict__ bih,
    const float* __restrict__ bhh, float* __restrict__ hout0,
    float* __restrict__ hout1, int nnodes) {
  constexpr int FH = 4 * H;
  constexpr int NTILES = FH / 16;    // N-tiles of the gate matrix
  constexpr int NT_PW = NTILES / 8;  // per wave (8 waves/block)
  constexpr int KCX = CIN / 32;      // K chunks over input features
  constexpr int KCH = H / 32;        // K chunks over hidden state
  constexpr int C4 = CIN / 4;

  const int set = blockIdx.y;
  const int* __restrict__ sfrm = set ? sfrm1 : sfrm0;
  const int* __restrict__ starts = set ? st1 : st0;
  const int* __restrict__ cnt = set ? cnt1 : cnt0;
  const int* __restrict__ mdeg = set ? md1 : md0;
  float* __restrict__ hout = set ? hout1 : hout0;

  __shared__ _Float16 Xs[16 * CIN];  // x_t tile (f16)
  __shared__ _Float16 Hs[16 * H];    // h_{t-1} tile (f16)
  __shared__ float Zs[16 * FH];      // gate pre-activations (f32)
  __shared__ float cS[16 * H];
  __shared__ float hS[16 * H];
  __shared__ float bS[FH];           // bih + bhh
  __shared__ int nStart[16];
  __shared__ int nCnt[16];

  const int tid = threadIdx.x;
  const int lane = tid & 31;
  const int wave = tid >> 5;
  const int m = lane & 15;     // row (A) / column (B) within 16
  const int hsel = lane >> 4;  // half-wave select
  const int tile = blockIdx.x;

  for (int i = tid; i < FH; i += 256) bS[i] = bih[i] + bhh[i];
  if (tid < 16) {
    int g = tile * 16 + tid;
    nStart[tid] = (g < nnodes) ? starts[g] : 0;
    nCnt[tid] = (g < nnodes) ? cnt[g] : 0;
  }
  for (int i = tid; i < 16 * H; i += 256) {
    cS[i] = 0.0f; hS[i] = 0.0f; Hs[i] = (_Float16)0.0f;
  }
  __syncthreads();

  // ---- preload B operands into registers --------------------------------
  // wave32 16-bit B layout: lane L -> column n = L&15;
// element e (0..15)     -> k = kchunk*32 + (L>>4)*8 + e + (e>>3)*8.
  v16h bX[NT_PW][KCX];
  v16h bH[NT_PW][KCH];
#pragma unroll
  for (int nt = 0; nt < NT_PW; ++nt) {
    const int row = (wave * NT_PW + nt) * 16 + m;  // gate column index
#pragma unroll
    for (int kc = 0; kc < KCX; ++kc) {
      const float* w = Wih + row * CIN + kc * 32 + hsel * 8;
#pragma unroll
      for (int e = 0; e < 16; ++e) bX[nt][kc][e] = (_Float16)w[e + (e >> 3) * 8];
    }
#pragma unroll
    for (int kc = 0; kc < KCH; ++kc) {
      const float* w = Whh + row * H + kc * 32 + hsel * 8;
#pragma unroll
      for (int e = 0; e < 16; ++e) bH[nt][kc][e] = (_Float16)w[e + (e >> 3) * 8];
    }
  }

  const v4h fill4 = {(_Float16)FILLV, (_Float16)FILLV, (_Float16)FILLV,
                     (_Float16)FILLV};

  const int T = *mdeg;  // uniform scalar load, no host sync needed
  for (int t = 0; t < T; ++t) {
    // gather x_t (padding rows get FILL, matching to_dense_batch semantics);
    // one 8B load + one 8B LDS store per thread.
    for (int i = tid; i < 16 * C4; i += 256) {
      int node = i / C4, k4 = i - node * C4;
      v4h v = fill4;
      if (t < nCnt[node]) {
        int src = sfrm[nStart[node] + t];
        v = *(const v4h*)(xsrc + src * CIN + k4 * 4);
        if (k4 == 0 && t + 1 < nCnt[node])  // prefetch next step's row
          __builtin_prefetch((const void*)(xsrc + sfrm[nStart[node] + t + 1] * CIN), 0, 1);
      }
      *(v4h*)(Xs + node * CIN + k4 * 4) = v;
    }
    __syncthreads();

    // Z[16 x FH] = X_t * Wih^T + H_{t-1} * Whh^T  (f32 accumulate)
#pragma unroll
    for (int nt = 0; nt < NT_PW; ++nt) {
      v8f acc = {};
#pragma unroll
      for (int kc = 0; kc < KCX; ++kc) {
        // wave32 16-bit A layout: lane L -> row m = L&15;
        // elems 0..7  -> k = kchunk*32 + (L>>4)*8 + e
        // elems 8..15 -> k = kchunk*32 + 16 + (L>>4)*8 + e
        v16h a;
        const _Float16* p = Xs + m * CIN + kc * 32 + hsel * 8;
#pragma unroll
        for (int e = 0; e < 8; ++e) { a[e] = p[e]; a[8 + e] = p[16 + e]; }
        acc = __builtin_amdgcn_wmma_f32_16x16x32_f16(false, a, false,
                                                     bX[nt][kc], (short)0, acc,
                                                     false, false);
      }
#pragma unroll
      for (int kc = 0; kc < KCH; ++kc) {
        v16h a;
        const _Float16* p = Hs + m * H + kc * 32 + hsel * 8;
#pragma unroll
        for (int e = 0; e < 8; ++e) { a[e] = p[e]; a[8 + e] = p[16 + e]; }
        acc = __builtin_amdgcn_wmma_f32_16x16x32_f16(false, a, false,
                                                     bH[nt][kc], (short)0, acc,
                                                     false, false);
      }
      // C/D f32 layout: VGPR r, lane L -> M = r + 8*(L>>4), N = L&15
      const int n0 = (wave * NT_PW + nt) * 16;
#pragma unroll
      for (int r = 0; r < 8; ++r)
        Zs[(r + 8 * hsel) * FH + n0 + m] = acc[r];
    }
    __syncthreads();

    // gates: i,f,g,o chunks of Z; c/h update via native tanh (TRANS)
    for (int i = tid; i < 16 * H; i += 256) {
      int node = i / H, hh = i - node * H;
      const float* z = Zs + node * FH;
      float zi = z[hh] + bS[hh];
      float zf = z[H + hh] + bS[H + hh];
      float zg = z[2 * H + hh] + bS[2 * H + hh];
      float zo = z[3 * H + hh] + bS[3 * H + hh];
      float c = sig_(zf) * cS[i] + sig_(zi) * tanh_(zg);
      float h = sig_(zo) * tanh_(c);
      cS[i] = c; hS[i] = h; Hs[i] = (_Float16)h;
    }
    __syncthreads();
  }

  for (int i = tid; i < 16 * H; i += 256) {
    int node = i / H;
    if (tile * 16 + node < nnodes) hout[tile * (16 * H) + i] = hS[i];
  }
}

// ---------------------------- combine + MLP --------------------------------

// average edge sets + bias; f16 output feeds the next LSTM layer's gather
__global__ void k_combine_h(const float* __restrict__ a,
                            const float* __restrict__ b,
                            const float* __restrict__ bias,
                            _Float16* __restrict__ o, int total, int H) {
  int i = blockIdx.x * blockDim.x + threadIdx.x;
  if (i < total) o[i] = (_Float16)(0.5f * (a[i] + b[i]) + bias[i % H]);
}

// average edge sets + bias; f32 output feeds the FC head
__global__ void k_combine_f(const float* __restrict__ a,
                            const float* __restrict__ b,
                            const float* __restrict__ bias,
                            float* __restrict__ o, int total, int H) {
  int i = blockIdx.x * blockDim.x + threadIdx.x;
  if (i < total) o[i] = 0.5f * (a[i] + b[i]) + bias[i % H];
}

__global__ __launch_bounds__(128) void k_fc(
    const float* __restrict__ x, const float* __restrict__ W1,
    const float* __restrict__ b1, const float* __restrict__ W2,
    const float* __restrict__ b2, float* __restrict__ out, int nnodes) {
  __shared__ float hb[128];
  int node = blockIdx.x;
  int j = threadIdx.x;  // 0..127
  float acc = b1[j];
  const float* xv = x + node * 32;
  const float* w = W1 + j * 32;
#pragma unroll
  for (int k = 0; k < 32; ++k) acc += xv[k] * w[k];
  hb[j] = fmaxf(acc, 0.0f);
  __syncthreads();
  if (j < 2) {
    float o = b2[j];
    const float* w2 = W2 + j * 128;
    for (int k = 0; k < 128; ++k) o += hb[k] * w2[k];
    out[node * 2 + j] = o;
  }
}

// ------------------------------ launcher -----------------------------------

extern "C" void kernel_launch(void* const* d_in, const int* in_sizes, int n_in,
                              void* d_out, int out_size, void* d_ws,
                              size_t ws_size, hipStream_t stream) {
  const float* x = (const float*)d_in[0];
  const float* ts = (const float*)d_in[1];
  const int* eA = (const int*)d_in[2];
  const int* eB = (const int*)d_in[3];
  const float* Wih0 = (const float*)d_in[4];
  const float* Whh0 = (const float*)d_in[5];
  const float* bih0 = (const float*)d_in[6];
  const float* bhh0 = (const float*)d_in[7];
  const float* bias0 = (const float*)d_in[8];
  const float* Wih1 = (const float*)d_in[9];
  const float* Whh1 = (const float*)d_in[10];
  const float* bih1 = (const float*)d_in[11];
  const float* bhh1 = (const float*)d_in[12];
  const float* bias1 = (const float*)d_in[13];
  const float* W1 = (const float*)d_in[14];
  const float* b1 = (const float*)d_in[15];
  const float* W2 = (const float*)d_in[16];
  const float* b2 = (const float*)d_in[17];

  const int Nn = in_sizes[1];     // 10000
  const int E = in_sizes[2] / 2;  // 160000

  // ---- carve workspace (~10 MB) ----
  char* p = (char*)d_ws;
  auto carve = [&](size_t bytes) -> void* {
    void* r = (void*)p;
    p += (bytes + 255) & ~(size_t)255;
    return r;
  };
  int* cntA = (int*)carve((size_t)Nn * 4);
  int* cntB = (int*)carve((size_t)Nn * 4);
  int* stA = (int*)carve((size_t)Nn * 4);
  int* stB = (int*)carve((size_t)Nn * 4);
  int* frmA = (int*)carve((size_t)E * 4);
  int* frmB = (int*)carve((size_t)E * 4);
  int* mdA = (int*)carve(4);
  int* mdB = (int*)carve(4);
  float* hb0 = (float*)carve((size_t)Nn * 64 * 4);
  float* hb1 = (float*)carve((size_t)Nn * 64 * 4);
  _Float16* xh = (_Float16*)carve((size_t)Nn * 64 * 2);
  _Float16* x1h = (_Float16*)carve((size_t)Nn * 64 * 2);
  float* x2 = (float*)carve((size_t)Nn * 32 * 4);
  (void)ws_size; (void)n_in; (void)out_size;

  const int nb = (Nn + 255) / 256;
  const int ebk = (E + 255) / 256;
  const int tiles = (Nn + 15) / 16;  // 625

  // ---- prep (both edge sets concurrently via grid.y) ----
  k_zero2<<<dim3(nb, 2), 256, 0, stream>>>(cntA, cntB, Nn);
  k_count2<<<dim3(ebk, 2), 256, 0, stream>>>(eA, eB, ts, cntA, cntB, E);
  k_scan2<<<2, 256, 0, stream>>>(cntA, cntB, stA, stB, mdA, mdB, Nn);
  k_fill2<<<dim3(nb, 2), 256, 0, stream>>>(eA, eB, ts, stA, stB, frmA, frmB,
                                           Nn, E);
  k_f32_to_f16<<<(Nn * 16 + 255) / 256, 256, 0, stream>>>(x, xh, Nn * 16);

  // ---- layer 0: C_IN=64 -> H=64 (both edge sets in one grid) ----
  k_lstm<64, 64><<<dim3(tiles, 2), 256, 0, stream>>>(
      xh, frmA, frmB, stA, stB, cntA, cntB, mdA, mdB, Wih0, Whh0, bih0, bhh0,
      hb0, hb1, Nn);
  k_combine_h<<<(Nn * 64 + 255) / 256, 256, 0, stream>>>(hb0, hb1, bias0, x1h,
                                                         Nn * 64, 64);

  // ---- layer 1: C=64 -> H=32 ----
  k_lstm<64, 32><<<dim3(tiles, 2), 256, 0, stream>>>(
      x1h, frmA, frmB, stA, stB, cntA, cntB, mdA, mdB, Wih1, Whh1, bih1, bhh1,
      hb0, hb1, Nn);
  k_combine_f<<<(Nn * 32 + 255) / 256, 256, 0, stream>>>(hb0, hb1, bias1, x2,
                                                         Nn * 32, 32);

  // ---- MLP head -> [N,2] f32 ----
  k_fc<<<Nn, 128, 0, stream>>>(x2, W1, b1, W2, b2, (float*)d_out, Nn);
}
